// TKFFTConv_78142634983460
// MI455X (gfx1250) — matrix operand order repaired and checked by reading, not compiled, not run on hardware
//
#include <hip/hip_runtime.h>

#define N1 32
#define HH 1024
#define BB 16
#define LSTR 40   // padded row stride (ushort): 80B rows -> 16B-aligned b128 LDS ops

typedef __attribute__((ext_vector_type(16))) __bf16 v16bf;
typedef __attribute__((ext_vector_type(8)))  float  v8f;

union Frag  { v16bf v; uint4 q[2]; unsigned int u[8]; unsigned short s[16]; };
union Pack8 { unsigned short s[8]; uint4 q; };

__device__ inline unsigned short f2bf(float f) {       // native RNE f32 -> bf16
  union { __bf16 b; unsigned short s; } u;
  u.b = (__bf16)f;
  return u.s;
}
__device__ inline float bf2f(unsigned short s) {
  return __uint_as_float(((unsigned int)s) << 16);
}

__device__ inline v8f wmma_bf16(v16bf a, v16bf b, v8f c) {
  return __builtin_amdgcn_wmma_f32_16x16x32_bf16(false, a, false, b, (short)0, c,
                                                 false, false);
}

// A-operand (16x32) fragment for rows [16*mi,+16): elements i in [0,16) are
// K = hk*16 + i  -> two aligned b128 loads.
__device__ inline v16bf ldsA(const unsigned short* m, int mi, int lane) {
  const int lm = lane & 15, hk = lane >> 4;
  Frag r;
  const unsigned short* row = m + (16 * mi + lm) * LSTR + hk * 16;
  r.q[0] = *(const uint4*)(row);
  r.q[1] = *(const uint4*)(row + 8);
  return r.v;
}

// B-operand (32x16) fragment from a TRANSPOSED matrix (mt[col][row]):
// K = (lane>>4)*16 + i, N = lane&15 -> 16 contiguous u16 -> two b128 loads.
__device__ inline v16bf ldsBT(const unsigned short* mt, int ni, int lane, bool neg) {
  const int lm = lane & 15, hk = lane >> 4;
  Frag r;
  const unsigned short* row = mt + (16 * ni + lm) * LSTR + hk * 16;
  r.q[0] = *(const uint4*)(row);
  r.q[1] = *(const uint4*)(row + 8);
  if (neg)
#pragma unroll
    for (int p = 0; p < 8; ++p) r.u[p] ^= 0x80008000u;
  return r.v;
}

__device__ inline v16bf negf(v16bf a) {   // negate packed bf16 (sign-bit xor)
  Frag t; t.v = a;
#pragma unroll
  for (int i = 0; i < 8; ++i) t.u[i] ^= 0x80008000u;
  return t.v;
}

// ------- Kernel 1: kfT[h] = FFT_1024(k[h]), stored TRANSPOSED [h][col][row] --------
__global__ __launch_bounds__(128) void kfft_pre(
    const float* __restrict__ k,
    const unsigned short* __restrict__ f_re, const unsigned short* __restrict__ f_im,
    const unsigned short* __restrict__ tw_re, const unsigned short* __restrict__ tw_im,
    unsigned short* __restrict__ kf) {
  __shared__ __align__(16) unsigned short cstA[2][32 * LSTR];  // Fre,Fim (row-major)
  __shared__ __align__(16) unsigned short cstT[2][32 * LSTR];  // Fre^T,Fim^T
  __shared__ __align__(16) unsigned short twT[2][32 * LSTR];   // tw^T re,im
  __shared__ __align__(16) unsigned short xch[4][2][32 * LSTR];

  for (int i = threadIdx.x; i < 1024; i += 128) {
    int r = i >> 5, c = i & 31, d = r * LSTR + c, dt = c * LSTR + r;
    unsigned short fr = f_re[i], fi = f_im[i];
    cstA[0][d] = fr;  cstA[1][d] = fi;
    cstT[0][dt] = fr; cstT[1][dt] = fi;
    twT[0][dt] = tw_re[i]; twT[1][dt] = tw_im[i];
  }
  __syncthreads();

  const int w = threadIdx.x >> 5, lane = threadIdx.x & 31;
  const int lm = lane & 15, hk = lane >> 4;
  const int h = blockIdx.x * 4 + w;
  const float* kt = k + (size_t)h * 1024;

  v16bf kb[2];
#pragma unroll
  for (int ni = 0; ni < 2; ++ni) {
    Frag r;
#pragma unroll
    for (int i = 0; i < 16; ++i)
      r.s[i] = f2bf(kt[(hk * 16 + i) * 32 + 16 * ni + lm]);
    kb[ni] = r.v;
  }

  v8f xr[2][2], xi[2][2];
#pragma unroll
  for (int mi = 0; mi < 2; ++mi) {
    v16bf aFr = ldsA(cstA[0], mi, lane), aFi = ldsA(cstA[1], mi, lane);
#pragma unroll
    for (int ni = 0; ni < 2; ++ni) {
      v8f z = {};
      xr[mi][ni] = wmma_bf16(aFr, kb[ni], z);
      xi[mi][ni] = wmma_bf16(aFi, kb[ni], z);
    }
  }

  unsigned short* XR = xch[w][0];
  unsigned short* XI = xch[w][1];
#pragma unroll
  for (int mi = 0; mi < 2; ++mi)
#pragma unroll
    for (int ni = 0; ni < 2; ++ni) {
      int col = 16 * ni + lm;
      int tb = col * LSTR + 16 * mi + 8 * hk;      // transposed, contiguous in r
      Pack8 tpr, tpi;
      tpr.q = *(const uint4*)(&twT[0][tb]);
      tpi.q = *(const uint4*)(&twT[1][tb]);
#pragma unroll
      for (int r = 0; r < 8; ++r) {
        int row = 16 * mi + 8 * hk + r, d = row * LSTR + col;
        float tr = bf2f(tpr.s[r]) * 1024.0f;   // tw holds exp()/N; N*tw exact in bf16
        float ti = bf2f(tpi.s[r]) * 1024.0f;
        float a = xr[mi][ni][r], c = xi[mi][ni][r];
        XR[d] = f2bf(a * tr - c * ti);
        XI[d] = f2bf(a * ti + c * tr);
      }
    }

  v16bf Fbr[2] = { ldsBT(cstT[0], 0, lane, false), ldsBT(cstT[0], 1, lane, false) };
  v16bf Fbi[2] = { ldsBT(cstT[1], 0, lane, false), ldsBT(cstT[1], 1, lane, false) };
#pragma unroll
  for (int mi = 0; mi < 2; ++mi) {
    v16bf ar = ldsA(XR, mi, lane), ai = ldsA(XI, mi, lane), an = negf(ai);
#pragma unroll
    for (int ni = 0; ni < 2; ++ni) {
      v8f z = {};
      xr[mi][ni] = wmma_bf16(ar, Fbr[ni], wmma_bf16(an, Fbi[ni], z));
      xi[mi][ni] = wmma_bf16(ar, Fbi[ni], wmma_bf16(ai, Fbr[ni], z));
    }
  }

  // store transposed: kf[h][col][row], one b128 per (mi,ni,component)
  unsigned short* kre = kf + (size_t)h * 1024;
  unsigned short* kim = kre + (size_t)HH * 1024;
#pragma unroll
  for (int mi = 0; mi < 2; ++mi)
#pragma unroll
    for (int ni = 0; ni < 2; ++ni) {
      int base = (16 * ni + lm) * 32 + 16 * mi + 8 * hk;
      Pack8 pr, pi;
#pragma unroll
      for (int r = 0; r < 8; ++r) {
        pr.s[r] = f2bf(xr[mi][ni][r]);
        pi.s[r] = f2bf(xi[mi][ni][r]);
      }
      *(uint4*)(kre + base) = pr.q;
      *(uint4*)(kim + base) = pi.q;
    }
}

// ---------------- Kernel 2: main conv, one wave per (b,h) pair ----------------------
__global__ __launch_bounds__(128) void fftconv_main(
    const float* __restrict__ u,
    const unsigned short* __restrict__ f_re,  const unsigned short* __restrict__ f_im,
    const unsigned short* __restrict__ fi_re, const unsigned short* __restrict__ fi_im,
    const unsigned short* __restrict__ tw_re, const unsigned short* __restrict__ tw_im,
    const unsigned short* __restrict__ twi_re,const unsigned short* __restrict__ twi_im,
    const unsigned short* __restrict__ kf,
    float* __restrict__ out) {
  __shared__ __align__(16) unsigned short cstA[4][32 * LSTR]; // F,Finv re/im (row-major)
  __shared__ __align__(16) unsigned short cstT[4][32 * LSTR]; // F^T, Finv^T re/im
  __shared__ __align__(16) unsigned short twT[4][32 * LSTR];  // tw^T, twinv^T re/im
  __shared__ __align__(16) unsigned short xch[4][2][32 * LSTR];

  for (int i = threadIdx.x; i < 1024; i += 128) {
    int r = i >> 5, c = i & 31, d = r * LSTR + c, dt = c * LSTR + r;
    unsigned short fr = f_re[i],  fi = f_im[i];
    unsigned short gr = fi_re[i], gi = fi_im[i];
    cstA[0][d] = fr;  cstA[1][d] = fi;
    cstA[2][d] = gr;  cstA[3][d] = gi;
    cstT[0][dt] = fr; cstT[1][dt] = fi;
    cstT[2][dt] = gr; cstT[3][dt] = gi;
    twT[0][dt] = tw_re[i];  twT[1][dt] = tw_im[i];
    twT[2][dt] = twi_re[i]; twT[3][dt] = twi_im[i];
  }
  __syncthreads();

  const int w = threadIdx.x >> 5, lane = threadIdx.x & 31;
  const int lm = lane & 15, hk = lane >> 4;
  const int pair = blockIdx.x * 4 + w;       // == b*H + h
  const int h = pair & (HH - 1);
  const float* ut = u + (size_t)pair * 1024;

  // u tile -> bf16 B-operand fragments (built directly from global, coalesced b32)
  v16bf ub[2];
#pragma unroll
  for (int ni = 0; ni < 2; ++ni) {
    Frag r;
#pragma unroll
    for (int i = 0; i < 16; ++i)
      r.s[i] = f2bf(ut[(hk * 16 + i) * 32 + 16 * ni + lm]);
    ub[ni] = r.v;
  }

  v8f xr[2][2], xi[2][2];
  // X = F @ A  (F complex, A real): 8 WMMA
#pragma unroll
  for (int mi = 0; mi < 2; ++mi) {
    v16bf aFr = ldsA(cstA[0], mi, lane), aFi = ldsA(cstA[1], mi, lane);
#pragma unroll
    for (int ni = 0; ni < 2; ++ni) {
      v8f z = {};
      xr[mi][ni] = wmma_bf16(aFr, ub[ni], z);
      xi[mi][ni] = wmma_bf16(aFi, ub[ni], z);
    }
  }

  unsigned short* XR = xch[w][0];
  unsigned short* XI = xch[w][1];

  // ∘ tw (vector b128 reads from tw^T) -> exchange LDS (row-major; A-operands next)
#pragma unroll
  for (int mi = 0; mi < 2; ++mi)
#pragma unroll
    for (int ni = 0; ni < 2; ++ni) {
      int col = 16 * ni + lm;
      int tb = col * LSTR + 16 * mi + 8 * hk;
      Pack8 tpr, tpi;
      tpr.q = *(const uint4*)(&twT[0][tb]);
      tpi.q = *(const uint4*)(&twT[1][tb]);
#pragma unroll
      for (int r = 0; r < 8; ++r) {
        int row = 16 * mi + 8 * hk + r, d = row * LSTR + col;
        float tr = bf2f(tpr.s[r]), ti = bf2f(tpi.s[r]);
        float a = xr[mi][ni][r], c = xi[mi][ni][r];
        XR[d] = f2bf(a * tr - c * ti);
        XI[d] = f2bf(a * ti + c * tr);
      }
    }

  // X2 = X @ F  (complex x complex): 16 WMMA
  {
    v16bf Fbr[2] = { ldsBT(cstT[0], 0, lane, false), ldsBT(cstT[0], 1, lane, false) };
    v16bf Fbi[2] = { ldsBT(cstT[1], 0, lane, false), ldsBT(cstT[1], 1, lane, false) };
#pragma unroll
    for (int mi = 0; mi < 2; ++mi) {
      v16bf ar = ldsA(XR, mi, lane), ai = ldsA(XI, mi, lane), an = negf(ai);
#pragma unroll
      for (int ni = 0; ni < 2; ++ni) {
        v8f z = {};
        xr[mi][ni] = wmma_bf16(ar, Fbr[ni], wmma_bf16(an, Fbi[ni], z));
        xi[mi][ni] = wmma_bf16(ar, Fbi[ni], wmma_bf16(ai, Fbr[ni], z));
      }
    }
  }

  // ∘ kfT (transposed in ws -> b128 loads) -> exchange LDS (row-major, A-operands)
  {
    const unsigned short* kre = kf + (size_t)h * 1024;
    const unsigned short* kim = kre + (size_t)HH * 1024;
#pragma unroll
    for (int mi = 0; mi < 2; ++mi)
#pragma unroll
      for (int ni = 0; ni < 2; ++ni) {
        int col = 16 * ni + lm;
        int base = col * 32 + 16 * mi + 8 * hk;
        Pack8 pr, pi;
        pr.q = *(const uint4*)(kre + base);
        pi.q = *(const uint4*)(kim + base);
#pragma unroll
        for (int r = 0; r < 8; ++r) {
          int row = 16 * mi + 8 * hk + r;
          float kr = bf2f(pr.s[r]), ki = bf2f(pi.s[r]);
          float a = xr[mi][ni][r], c = xi[mi][ni][r];
          XR[row * LSTR + col] = f2bf(a * kr - c * ki);
          XI[row * LSTR + col] = f2bf(a * ki + c * kr);
        }
      }
  }

  // Z = Y @ Finv: 16 WMMA
  {
    v16bf Gbr[2] = { ldsBT(cstT[2], 0, lane, false), ldsBT(cstT[2], 1, lane, false) };
    v16bf Gbi[2] = { ldsBT(cstT[3], 0, lane, false), ldsBT(cstT[3], 1, lane, false) };
#pragma unroll
    for (int mi = 0; mi < 2; ++mi) {
      v16bf ar = ldsA(XR, mi, lane), ai = ldsA(XI, mi, lane), an = negf(ai);
#pragma unroll
      for (int ni = 0; ni < 2; ++ni) {
        v8f z = {};
        xr[mi][ni] = wmma_bf16(ar, Gbr[ni], wmma_bf16(an, Gbi[ni], z));
        xi[mi][ni] = wmma_bf16(ar, Gbi[ni], wmma_bf16(ai, Gbr[ni], z));
      }
    }
  }

  // ∘ twinv (vector b128 reads) -> exchange LDS stored TRANSPOSED, packed b128 stores
#pragma unroll
  for (int mi = 0; mi < 2; ++mi)
#pragma unroll
    for (int ni = 0; ni < 2; ++ni) {
      int col = 16 * ni + lm;
      int tb = col * LSTR + 16 * mi + 8 * hk;
      Pack8 tpr, tpi, pr, pi;
      tpr.q = *(const uint4*)(&twT[2][tb]);
      tpi.q = *(const uint4*)(&twT[3][tb]);
#pragma unroll
      for (int r = 0; r < 8; ++r) {
        float tr = bf2f(tpr.s[r]), ti = bf2f(tpi.s[r]);
        float a = xr[mi][ni][r], c = xi[mi][ni][r];
        pr.s[r] = f2bf(a * tr - c * ti);
        pi.s[r] = f2bf(a * ti + c * tr);
      }
      *(uint4*)(XR + tb) = pr.q;
      *(uint4*)(XI + tb) = pi.q;
    }

  // out = Re(Finv @ Z): 8 WMMA, coalesced f32 stores
  {
    v16bf bZr[2] = { ldsBT(XR, 0, lane, false), ldsBT(XR, 1, lane, false) };
    v16bf bZn[2] = { ldsBT(XI, 0, lane, true),  ldsBT(XI, 1, lane, true) }; // -Im(Z)
    float* ot = out + (size_t)pair * 1024;
#pragma unroll
    for (int mi = 0; mi < 2; ++mi) {
      v16bf air = ldsA(cstA[2], mi, lane), aii = ldsA(cstA[3], mi, lane);
#pragma unroll
      for (int ni = 0; ni < 2; ++ni) {
        v8f z = {};
        v8f o = wmma_bf16(air, bZr[ni], wmma_bf16(aii, bZn[ni], z));
#pragma unroll
        for (int r = 0; r < 8; ++r) {
          int row = 16 * mi + 8 * hk + r, col = 16 * ni + lm;
          ot[row * 32 + col] = o[r];
        }
      }
    }
  }
}

extern "C" void kernel_launch(void* const* d_in, const int* in_sizes, int n_in,
                              void* d_out, int out_size, void* d_ws, size_t ws_size,
                              hipStream_t stream) {
  const float* u  = (const float*)d_in[0];
  const float* k  = (const float*)d_in[1];
  const unsigned short* f_re   = (const unsigned short*)d_in[2];
  const unsigned short* f_im   = (const unsigned short*)d_in[3];
  const unsigned short* fi_re  = (const unsigned short*)d_in[4];
  const unsigned short* fi_im  = (const unsigned short*)d_in[5];
  const unsigned short* tw_re  = (const unsigned short*)d_in[6];
  const unsigned short* tw_im  = (const unsigned short*)d_in[7];
  const unsigned short* twi_re = (const unsigned short*)d_in[8];
  const unsigned short* twi_im = (const unsigned short*)d_in[9];
  unsigned short* kf = (unsigned short*)d_ws;   // [2][H][32][32] bf16 transposed tiles

  kfft_pre<<<HH / 4, 128, 0, stream>>>(k, f_re, f_im, tw_re, tw_im, kf);
  fftconv_main<<<(BB * HH) / 4, 128, 0, stream>>>(
      u, f_re, f_im, fi_re, fi_im, tw_re, tw_im, twi_re, twi_im, kf,
      (float*)d_out);
}
